// TrittentionTriton_36043365548472
// MI455X (gfx1250) — matrix operand order repaired and checked by reading, not compiled, not run on hardware
//
#include <hip/hip_runtime.h>

typedef __attribute__((ext_vector_type(16))) _Float16 v16h;
typedef __attribute__((ext_vector_type(8)))  float    v8f;
typedef __attribute__((ext_vector_type(4)))  float    f32x4;

#define NHEAD 8
#define SEQ   256
#define DIM   64
#define NW    8            // waves per block (wave32)
#define NTHR  (NW * 32)
#define SCALE 0.125f
#define NEGINF (-1e30f)

// LDS layout (bytes)
#define OFF_K1   0          // [SEQ][DIM] f16          (32 KB)
#define OFF_SV1T 32768      // [DIM][SEQ] f16 silu^T   (32 KB)
#define OFF_K2   65536      // [SEQ][DIM] f32          (64 KB)
#define OFF_P    131072     // [NW][16][SEQ] f16       (64 KB)
#define SMEM_SZ  196608

// One block per (query-tile of 16, head). 8 waves split the k (second-key)
// loop with stride 8; each wave runs a private flash-style online softmax over
// (j,k) pairs and the partials are merged through LDS at the end.
__global__ __launch_bounds__(NTHR) void tritt_fwd(
    const float* __restrict__ qg_, const float* __restrict__ k1g_,
    const float* __restrict__ k2g_, const float* __restrict__ v1g_,
    const float* __restrict__ v2g_, float* __restrict__ out)
{
  extern __shared__ char smem[];
  _Float16* k1h  = (_Float16*)(smem + OFF_K1);
  _Float16* sv1t = (_Float16*)(smem + OFF_SV1T);
  float*    k2f  = (float*)   (smem + OFF_K2);
  _Float16* Pb   = (_Float16*)(smem + OFF_P);

  const int head = blockIdx.y;
  const int t    = blockIdx.x;        // query tile index (16 rows)
  const int tid  = threadIdx.x;
  const int wave = tid >> 5;
  const int lane = tid & 31;
  const int l15  = lane & 15;
  const int half = lane >> 4;

  const float* qg  = qg_  + head * SEQ * DIM;
  const float* k1g = k1g_ + head * SEQ * DIM;
  const float* k2g = k2g_ + head * SEQ * DIM;
  const float* v1g = v1g_ + head * SEQ * DIM;
  const float* v2g = v2g_ + head * SEQ * DIM;

  // ---- stage K1 (f16), K2 (f32), silu(V1)^T (f16) into LDS ----
  {
    const f32x4* k14 = (const f32x4*)k1g;
    const f32x4* k24 = (const f32x4*)k2g;
    const f32x4* v14 = (const f32x4*)v1g;
    f32x4* k2s = (f32x4*)k2f;
    for (int idx = tid; idx < SEQ * DIM / 4; idx += NTHR) {
      f32x4 a = k14[idx];
      _Float16* d = k1h + idx * 4;
      d[0] = (_Float16)a.x; d[1] = (_Float16)a.y;
      d[2] = (_Float16)a.z; d[3] = (_Float16)a.w;
      k2s[idx] = k24[idx];
      f32x4 b = v14[idx];
      int j  = (idx * 4) >> 6;      // source row
      int d0 = (idx * 4) & 63;      // source col
      sv1t[(d0 + 0) * SEQ + j] = (_Float16)(b.x / (1.f + __expf(-b.x)));
      sv1t[(d0 + 1) * SEQ + j] = (_Float16)(b.y / (1.f + __expf(-b.y)));
      sv1t[(d0 + 2) * SEQ + j] = (_Float16)(b.z / (1.f + __expf(-b.z)));
      sv1t[(d0 + 3) * SEQ + j] = (_Float16)(b.w / (1.f + __expf(-b.w)));
    }
  }
  // zero this wave's P buffer once (pad j-tile for odd counts stays zero)
  {
    uint32_t* pz = (uint32_t*)(Pb + wave * 16 * SEQ);
    for (int idx = lane; idx < 16 * SEQ / 2; idx += 32) pz[idx] = 0u;
  }
  __syncthreads();

  // ---- per-lane Q in WMMA A-layout: M = lane&15, K(e)=e+16*half
  float qlo[16], qhi[16];
  {
    const int qrow = t * 16 + l15;
#pragma unroll
    for (int e = 0; e < 16; ++e) {
      int kd = e + (half << 4);
      qlo[e] = qg[qrow * DIM + kd];
      qhi[e] = qg[qrow * DIM + kd + 32];
    }
  }

  float m_run[8], l_run[8];
  v8f acc[4];
#pragma unroll
  for (int r = 0; r < 8; ++r) { m_run[r] = -INFINITY; l_run[r] = 0.f; }
#pragma unroll
  for (int dt = 0; dt < 4; ++dt) { v8f z = {}; acc[dt] = z; }

  const int imax = t * 16 + 15;
  const int nk   = imax + 1;        // causal: k <= i_max of tile
  const int njt  = t + 1;           // active j tiles (j <= i_max)
  const int ncj  = (njt + 1) >> 1;  // 32-wide K chunks for the PV GEMM
  _Float16* Pw = Pb + wave * 16 * SEQ;

  for (int k = wave; k < nk; k += NW) {
    // A = (q .* k2[k]) as f16, split over d: [0,32) and [32,64)
    v16h aLo, aHi;
#pragma unroll
    for (int e = 0; e < 16; ++e) {
      int kd = e + (half << 4);
      aLo[e] = (_Float16)(qlo[e] * k2f[k * DIM + kd]);
      aHi[e] = (_Float16)(qhi[e] * k2f[k * DIM + kd + 32]);
    }

    // Scores S[i,j] = A @ K1^T, per 16-wide j tile (two K=32 WMMAs each).
    v8f sc[16];
#pragma unroll
    for (int jt = 0; jt < 16; ++jt) {
      if (jt < njt) {   // wave-uniform branch (njt from blockIdx)
        const _Float16* bb = k1h + (jt * 16 + l15) * DIM + half * 16;
        v16h b0 = *(const v16h*)(bb);        // d =  half*16 + e
        v16h b1 = *(const v16h*)(bb + 32);   // d = 32 + half*16 + e
        v8f c = {};
        c = __builtin_amdgcn_wmma_f32_16x16x32_f16(false, aLo, false, b0,
                                                   (short)0, c, false, false);
        c = __builtin_amdgcn_wmma_f32_16x16x32_f16(false, aHi, false, b1,
                                                   (short)0, c, false, false);
        sc[jt] = c;
      }
    }

    // scale + causal mask + per-row max (C layout: M = r + 8*half, N = l15)
    float rmax[8];
#pragma unroll
    for (int r = 0; r < 8; ++r) rmax[r] = -INFINITY;
#pragma unroll
    for (int jt = 0; jt < 16; ++jt) {
      if (jt < njt) {
        int j = jt * 16 + l15;
#pragma unroll
        for (int r = 0; r < 8; ++r) {
          int i = t * 16 + r + 8 * half;
          float s = sc[jt][r] * SCALE;
          s = ((j <= i) && (k <= i)) ? s : NEGINF;
          sc[jt][r] = s;
          rmax[r] = fmaxf(rmax[r], s);
        }
      }
    }
#pragma unroll
    for (int r = 0; r < 8; ++r) {
#pragma unroll
      for (int mm = 1; mm < 16; mm <<= 1)
        rmax[r] = fmaxf(rmax[r], __shfl_xor(rmax[r], mm, 32));
    }

    float rs[8];
#pragma unroll
    for (int r = 0; r < 8; ++r) {
      float nm    = fmaxf(m_run[r], rmax[r]);
      float alpha = __expf(m_run[r] - nm);   // 0 for -inf start / stale rows
      m_run[r] = nm;
      l_run[r] *= alpha;
#pragma unroll
      for (int dt = 0; dt < 4; ++dt) acc[dt][r] *= alpha;
      rs[r] = 0.f;
    }

    // P = exp(S - m) -> f16 LDS, row-major [M][j] (matches A-layout reloads)
#pragma unroll
    for (int jt = 0; jt < 16; ++jt) {
      if (jt < njt) {
#pragma unroll
        for (int r = 0; r < 8; ++r) {
          float p = __expf(sc[jt][r] - m_run[r]);
          rs[r] += p;
          Pw[(r + 8 * half) * SEQ + jt * 16 + l15] = (_Float16)p;
        }
      }
    }
#pragma unroll
    for (int r = 0; r < 8; ++r) {
      float s = rs[r];
#pragma unroll
      for (int mm = 1; mm < 16; mm <<= 1) s += __shfl_xor(s, mm, 32);
      l_run[r] += s;
    }

    // acc[i,d] += (P @ silu(V1))[i,d] * v2[k,d]
#pragma unroll
    for (int dt = 0; dt < 4; ++dt) {
      v8f u = {};
      for (int jc = 0; jc < ncj; ++jc) {       // wave-uniform bound
        const _Float16* pr = Pw + l15 * SEQ + jc * 32 + half * 16;
        v16h pa;
#pragma unroll
        for (int e = 0; e < 8; ++e) { pa[e] = pr[e]; pa[e + 8] = pr[e + 8]; }
        v16h bv = *(const v16h*)(sv1t + (dt * 16 + l15) * SEQ + jc * 32 + half * 16);
        u = __builtin_amdgcn_wmma_f32_16x16x32_f16(false, pa, false, bv,
                                                   (short)0, u, false, false);
      }
      float v2v = v2g[k * DIM + dt * 16 + l15];
#pragma unroll
      for (int r = 0; r < 8; ++r) acc[dt][r] += u[r] * v2v;
    }
  }

  // ---- cross-wave softmax merge (reuse P region after barrier) ----
  __syncthreads();
  float* accL = (float*)(smem + OFF_P);                  // [NW][16][DIM] 32 KB
  float* mL   = (float*)(smem + OFF_P + NW * 16 * DIM * 4);  // [NW][16]
  float* lL   = mL + NW * 16;                            // [NW][16]
  if (l15 == 0) {
#pragma unroll
    for (int r = 0; r < 8; ++r) {
      mL[wave * 16 + r + 8 * half] = m_run[r];
      lL[wave * 16 + r + 8 * half] = l_run[r];
    }
  }
#pragma unroll
  for (int dt = 0; dt < 4; ++dt)
#pragma unroll
    for (int r = 0; r < 8; ++r)
      accL[(wave * 16 + r + 8 * half) * DIM + dt * 16 + l15] = acc[dt][r];
  __syncthreads();

  float* outO = out;                       // [NHEAD][SEQ][DIM]
  float* outL = out + NHEAD * SEQ * DIM;   // [NHEAD][SEQ]
  for (int e = tid; e < 16 * DIM; e += NTHR) {
    int row = e >> 6, d = e & 63;
    float M2 = -INFINITY;
#pragma unroll
    for (int w = 0; w < NW; ++w) M2 = fmaxf(M2, mL[w * 16 + row]);
    float Lsum = 0.f, num = 0.f;
#pragma unroll
    for (int w = 0; w < NW; ++w) {
      float wgt = __expf(mL[w * 16 + row] - M2);
      Lsum += lL[w * 16 + row] * wgt;
      num  += accL[(w * 16 + row) * DIM + d] * wgt;
    }
    outO[(head * SEQ + t * 16 + row) * DIM + d] = num / Lsum;
    if (d == 0) outL[head * SEQ + t * 16 + row] = M2 + __logf(Lsum);
  }
}

extern "C" void kernel_launch(void* const* d_in, const int* in_sizes, int n_in,
                              void* d_out, int out_size, void* d_ws, size_t ws_size,
                              hipStream_t stream) {
  (void)in_sizes; (void)n_in; (void)d_ws; (void)ws_size; (void)out_size;
  const float* q  = (const float*)d_in[0];
  const float* k1 = (const float*)d_in[1];
  const float* k2 = (const float*)d_in[2];
  const float* v1 = (const float*)d_in[3];
  const float* v2 = (const float*)d_in[4];
  float* out = (float*)d_out;
  dim3 grid(SEQ / 16, NHEAD);   // 16 query tiles x 8 heads
  dim3 block(NTHR);             // 8 wave32 waves
  tritt_fwd<<<grid, block, SMEM_SZ, stream>>>(q, k1, k2, v1, v2, out);
}
